// RN_70884140253769
// MI455X (gfx1250) — matrix operand order repaired and checked by reading, not compiled, not run on hardware
//
#include <hip/hip_runtime.h>
#include <hip/hip_bf16.h>

// ---------------------------------------------------------------------------
// Problem constants (from reference)
// ---------------------------------------------------------------------------
#define BATCH   4
#define NMAX    128
#define NPOOL   16
#define I_CNN   2048
#define QDIM    1024
#define LIN     1024
#define NCLS    16
#define D_BOX   (I_CNN + 4)          // 2052
#define D_IN    (2 * D_BOX + QDIM)   // 5128

// CDNA5 WMMA vector types (wave32, 16x16x32 bf16 -> f32)
typedef __attribute__((ext_vector_type(16))) __bf16        v16bf;
typedef __attribute__((ext_vector_type(8)))  float         v8f;
typedef __attribute__((ext_vector_type(4)))  unsigned int  v4u;

union Frag {
    v4u   q[2];   // two 16B chunks
    v16bf v;      // WMMA operand
};

__device__ __forceinline__ unsigned short f2bf(float f) {
    unsigned int u = __float_as_uint(f);
    u += 0x7fffu + ((u >> 16) & 1u);   // round-to-nearest-even
    return (unsigned short)(u >> 16);
}

// ---------------------------------------------------------------------------
// Workspace layout (bytes)
// ---------------------------------------------------------------------------
#define OFF_WP1   (size_t)0           // g1_w2 packed bf16   2 MB
#define OFF_WP2   (size_t)2097152     // g1_w3 packed
#define OFF_WP3   (size_t)4194304     // g2_w2 packed
#define OFF_WP4   (size_t)6291456     // g2_w3 packed
#define OFF_A1    (size_t)8388608     // [4,128,1024] f32
#define OFF_B1    (size_t)10485760
#define OFF_A2    (size_t)12582912
#define OFF_B2P   (size_t)14680064    // [4,16,1024]  f32
#define OFF_HQ1   (size_t)14942208    // [4,1024] f32
#define OFF_HQ2   (size_t)14958592
#define OFF_S1    (size_t)14974976    // [4,1024] f32 accum (zeroed)
#define OFF_S2    (size_t)14991360
#define OFF_RNO   (size_t)15007744
#define OFF_RNOB  (size_t)15024128
#define OFF_Z     (size_t)15040512

// ---------------------------------------------------------------------------
// 1) Pack four 1024x1024 f32 weights into WMMA-B-fragment bf16 layout.
//    B-frag (32x16, bf16): lane<16 holds K=kbase+{0..7,16..23} of column n=lane,
//    lane>=16 holds K=kbase+{8..15,24..31} of column n=lane-16. Each lane's 16 halfs
//    are stored contiguously (32B) so the GEMM loads two global b128s.
// ---------------------------------------------------------------------------
__global__ __launch_bounds__(256)
void pack_w_kernel(const float* __restrict__ w0, const float* __restrict__ w1,
                   const float* __restrict__ w2, const float* __restrict__ w3,
                   unsigned short* __restrict__ dst)
{
    size_t gid = (size_t)blockIdx.x * 256 + threadIdx.x;   // 4 * 2^20 total
    int mat  = (int)(gid >> 20);
    unsigned int rem = (unsigned int)(gid & 0xFFFFFu);
    int e    = rem & 15;
    int lane = (rem >> 4) & 31;
    int kt   = (rem >> 9) & 31;
    int nt   = rem >> 14;                       // 0..63
    int k = kt * 32 + ((lane >= 16) ? 8 : 0) + e + ((e >= 8) ? 8 : 0);
    int n = nt * 16 + (lane & 15);
    const float* src = (mat == 0) ? w0 : (mat == 1) ? w1 : (mat == 2) ? w2 : w3;
    size_t doff = (size_t)mat * 1048576 + ((size_t)(nt * 32 + kt) * 32 + lane) * 16 + e;
    dst[doff] = f2bf(src[(size_t)k * LIN + n]);
}

// ---------------------------------------------------------------------------
// 2) Projection: out[b,r,:] = concat(feat[b,r,:2048], coords[.,r,:4]) @ W
// ---------------------------------------------------------------------------
__global__ __launch_bounds__(256)
void proj_kernel(const float* __restrict__ feats, const float* __restrict__ coords,
                 const float* __restrict__ W, float* __restrict__ out,
                 int nrows, int coords_batched)
{
    __shared__ float xs[D_BOX];
    int gid = blockIdx.x;
    int b = gid / nrows, r = gid % nrows;
    int tid = threadIdx.x;
    const float* f = feats + ((size_t)b * nrows + r) * I_CNN;
    for (int k = tid; k < I_CNN; k += 256) xs[k] = f[k];
    if (tid < 4) {
        size_t ci = ((coords_batched ? ((size_t)b * nrows + r) : (size_t)r) * 4) + tid;
        xs[I_CNN + tid] = coords[ci];
    }
    __syncthreads();
    float a0 = 0.f, a1 = 0.f, a2 = 0.f, a3 = 0.f;
    for (int k = 0; k < D_BOX; ++k) {
        float xv = xs[k];
        const float* wr = W + (size_t)k * LIN;
        a0 += xv * wr[tid];
        a1 += xv * wr[tid + 256];
        a2 += xv * wr[tid + 512];
        a3 += xv * wr[tid + 768];
    }
    float* o = out + ((size_t)b * nrows + r) * LIN;
    o[tid] = a0; o[tid + 256] = a1; o[tid + 512] = a2; o[tid + 768] = a3;
}

// ---------------------------------------------------------------------------
// 3) GEMV: y[b,:] = (relu?)(x[b,:] @ W + bias), K = 1024, 1024 outputs
// ---------------------------------------------------------------------------
__global__ __launch_bounds__(256)
void gemv_kernel(const float* __restrict__ x, const float* __restrict__ W,
                 const float* __restrict__ bias, float* __restrict__ y,
                 int relu_flag)
{
    __shared__ float xs[1024];
    int b = blockIdx.x >> 2, chunk = blockIdx.x & 3, tid = threadIdx.x;
    const float* xb = x + (size_t)b * 1024;
    for (int k = tid; k < 1024; k += 256) xs[k] = xb[k];
    __syncthreads();
    int col = chunk * 256 + tid;
    float s = bias[col];
    for (int k = 0; k < 1024; ++k) s += xs[k] * W[(size_t)k * LIN + col];
    if (relu_flag) s = s > 0.f ? s : 0.f;
    y[(size_t)b * 1024 + col] = s;
}

__global__ __launch_bounds__(256)
void zero_kernel(float* __restrict__ p, int n)
{
    int i = blockIdx.x * 256 + threadIdx.x;
    if (i < n) p[i] = 0.f;
}

// ---------------------------------------------------------------------------
// 4) Fused relational MLP tile kernel (the hot loop).
//    Per workgroup: 32 (i,j) rows. Build h0 = relu(A[i]+B[j]+hq) in LDS as
//    bf16, then h1 = relu(h0 @ W2 + b2) via WMMA, h2 = relu(h1 @ W3 + b3),
//    then masked row-sum atomically accumulated into outsum[b,:].
//    8 waves: wave = 128-column block; each wave owns BOTH 16-row M-tiles so
//    every B fragment load feeds two WMMAs (halves L2 weight traffic).
// ---------------------------------------------------------------------------
#define PROWS    32
#define PTHREADS 256
#define H0STR    1032    // padded row stride in halfs (bank-conflict-free)

__global__ __launch_bounds__(PTHREADS)
void pair_kernel(const float* __restrict__ Arow, const float* __restrict__ Brow,
                 const float* __restrict__ hq,
                 const unsigned short* __restrict__ WpA, const float* __restrict__ b2,
                 const unsigned short* __restrict__ WpB, const float* __restrict__ b3,
                 const int* __restrict__ index,
                 int ncols_shift, int check_j, float* __restrict__ outsum)
{
    extern __shared__ unsigned short smem[];
    unsigned short* h0 = smem;                                  // [32][1032] bf16
    unsigned short* h1 = smem + PROWS * H0STR;                  // [32][1032] bf16
    float* wrow = (float*)(smem + 2 * PROWS * H0STR);           // [32]

    const int b        = blockIdx.y;
    const int tileBase = blockIdx.x * PROWS;
    const int nvalid   = index[b];
    const int tid      = threadIdx.x;
    const int ncols    = 1 << ncols_shift;
    const int cmask    = ncols - 1;

    const float* A   = Arow + (size_t)b * NMAX * LIN;
    const float* Bv  = Brow + (size_t)b * ncols * LIN;
    const float* hqv = hq   + (size_t)b * LIN;

    // ---- Phase A: materialize h0 tile (bf16) + row validity weights ----
    for (int e = tid; e < PROWS * 1024; e += PTHREADS) {
        int r = e >> 10, c = e & 1023;
        int p = tileBase + r;
        int i = p >> ncols_shift, j = p & cmask;
        float v = A[(size_t)i * LIN + c] + Bv[(size_t)j * LIN + c] + hqv[c];
        v = v > 0.f ? v : 0.f;
        h0[r * H0STR + c] = f2bf(v);
    }
    if (tid < PROWS) {
        int p = tileBase + tid;
        int i = p >> ncols_shift, j = p & cmask;
        float w = (i < nvalid) ? 1.f : 0.f;
        if (check_j) w *= (j < nvalid) ? 1.f : 0.f;
        wrow[tid] = w;
    }
    __syncthreads();

    const int nb   = tid >> 5;       // wave id = 128-column block
    const int lane = tid & 31;
    const int half = lane >> 4;      // A-frag K-interleave half
    const int ln   = lane & 15;      // A row within tile / C column

    v8f acc0[8], acc1[8];            // M-tile 0 (rows 0-15) and 1 (rows 16-31)

    // ---- GEMM1: acc = h0 @ W2 ----
    for (int nt = 0; nt < 8; ++nt)
        for (int e = 0; e < 8; ++e) { acc0[nt][e] = 0.f; acc1[nt][e] = 0.f; }

    for (int kt = 0; kt < 32; ++kt) {
        Frag af0, af1;
        const unsigned short* ap0 = h0 + ln * H0STR        + kt * 32 + half * 8;
        const unsigned short* ap1 = h0 + (16 + ln) * H0STR + kt * 32 + half * 8;
        af0.q[0] = *(const v4u*)(ap0);        // K = kbase + {0..7 | 8..15}
        af0.q[1] = *(const v4u*)(ap0 + 16);   // K = kbase + 16 + {0..7 | 8..15}
        af1.q[0] = *(const v4u*)(ap1);
        af1.q[1] = *(const v4u*)(ap1 + 16);
        for (int nt = 0; nt < 8; ++nt) {
            int ntg = nb * 8 + nt;
            const unsigned short* bp = WpA + ((size_t)(ntg * 32 + kt) * 32 + lane) * 16;
            Frag bf;
            bf.q[0] = *(const v4u*)(bp);
            bf.q[1] = *(const v4u*)(bp + 8);
            acc0[nt] = __builtin_amdgcn_wmma_f32_16x16x32_bf16(
                false, af0.v, false, bf.v, (short)0, acc0[nt], false, false);
            acc1[nt] = __builtin_amdgcn_wmma_f32_16x16x32_bf16(
                false, af1.v, false, bf.v, (short)0, acc1[nt], false, false);
        }
    }

    // ---- epilogue 1: h1 = relu(acc + b2) -> LDS bf16 ----
    for (int nt = 0; nt < 8; ++nt) {
        int cb = nb * 128 + nt * 16 + ln;
        float bb = b2[cb];
        for (int vv = 0; vv < 8; ++vv) {
            int row = vv + half * 8;            // C layout: VGPR v -> rows v / v+8
            float x0 = acc0[nt][vv] + bb;
            float x1 = acc1[nt][vv] + bb;
            x0 = x0 > 0.f ? x0 : 0.f;
            x1 = x1 > 0.f ? x1 : 0.f;
            h1[row * H0STR + cb]        = f2bf(x0);
            h1[(16 + row) * H0STR + cb] = f2bf(x1);
        }
    }
    __syncthreads();

    // ---- GEMM2: acc = h1 @ W3 ----
    for (int nt = 0; nt < 8; ++nt)
        for (int e = 0; e < 8; ++e) { acc0[nt][e] = 0.f; acc1[nt][e] = 0.f; }

    for (int kt = 0; kt < 32; ++kt) {
        Frag af0, af1;
        const unsigned short* ap0 = h1 + ln * H0STR        + kt * 32 + half * 8;
        const unsigned short* ap1 = h1 + (16 + ln) * H0STR + kt * 32 + half * 8;
        af0.q[0] = *(const v4u*)(ap0);
        af0.q[1] = *(const v4u*)(ap0 + 16);
        af1.q[0] = *(const v4u*)(ap1);
        af1.q[1] = *(const v4u*)(ap1 + 16);
        for (int nt = 0; nt < 8; ++nt) {
            int ntg = nb * 8 + nt;
            const unsigned short* bp = WpB + ((size_t)(ntg * 32 + kt) * 32 + lane) * 16;
            Frag bf;
            bf.q[0] = *(const v4u*)(bp);
            bf.q[1] = *(const v4u*)(bp + 8);
            acc0[nt] = __builtin_amdgcn_wmma_f32_16x16x32_bf16(
                false, af0.v, false, bf.v, (short)0, acc0[nt], false, false);
            acc1[nt] = __builtin_amdgcn_wmma_f32_16x16x32_bf16(
                false, af1.v, false, bf.v, (short)0, acc1[nt], false, false);
        }
    }

    // ---- epilogue 2: masked row-sum of relu(acc + b3) -> global atomics ----
    float* gs = outsum + (size_t)b * LIN;
    for (int nt = 0; nt < 8; ++nt) {
        int cb = nb * 128 + nt * 16 + ln;
        float bb = b3[cb];
        float psum = 0.f;
        for (int vv = 0; vv < 8; ++vv) {
            int row = vv + half * 8;
            float x0 = acc0[nt][vv] + bb;
            float x1 = acc1[nt][vv] + bb;
            x0 = x0 > 0.f ? x0 : 0.f;
            x1 = x1 > 0.f ? x1 : 0.f;
            psum += wrow[row] * x0 + wrow[16 + row] * x1;
        }
        atomicAdd(&gs[cb], psum);   // lanes L and L+16 cover rows 0-7 / 8-15
    }
}

// ---------------------------------------------------------------------------
// 5) Final classifier head
// ---------------------------------------------------------------------------
__global__ __launch_bounds__(256)
void finalz_kernel(const float* __restrict__ rno, const float* __restrict__ rnob,
                   const float* __restrict__ W, const float* __restrict__ bias,
                   float* __restrict__ z)
{
    __shared__ float xs[2048];
    int b = blockIdx.x >> 2, chunk = blockIdx.x & 3, tid = threadIdx.x;
    for (int k = tid; k < 1024; k += 256) {
        xs[k]        = rno[(size_t)b * 1024 + k];
        xs[1024 + k] = rnob[(size_t)b * 1024 + k];
    }
    __syncthreads();
    int col = chunk * 256 + tid;
    float s = bias[col];
    for (int k = 0; k < 2048; ++k) s += xs[k] * W[(size_t)k * 1024 + col];
    s = s > 0.f ? s : 0.f;
    z[(size_t)b * 1024 + col] = s;
}

__global__ __launch_bounds__(256)
void finalout_kernel(const float* __restrict__ z, const float* __restrict__ W,
                     const float* __restrict__ bias, float* __restrict__ out)
{
    __shared__ float part[256];
    int b = blockIdx.x, tid = threadIdx.x;
    int c = tid & 15, g = tid >> 4;           // 16 k-groups x 16 cols
    float s = 0.f;
    for (int k = g * 64; k < g * 64 + 64; ++k)
        s += z[(size_t)b * 1024 + k] * W[(size_t)k * NCLS + c];
    part[tid] = s;
    __syncthreads();
    if (tid < NCLS) {
        float t = bias[tid];
        for (int g2 = 0; g2 < 16; ++g2) t += part[g2 * 16 + tid];
        out[(size_t)b * NCLS + tid] = t;
    }
}

// ---------------------------------------------------------------------------
// Launch
// ---------------------------------------------------------------------------
extern "C" void kernel_launch(void* const* d_in, const int* in_sizes, int n_in,
                              void* d_out, int out_size, void* d_ws, size_t ws_size,
                              hipStream_t stream)
{
    const float* pooled     = (const float*)d_in[1];
    const float* box_feats  = (const float*)d_in[2];
    const float* q_feats    = (const float*)d_in[3];
    const float* box_coords = (const float*)d_in[4];
    const int*   index      = (const int*)  d_in[5];
    const float* g1_w1 = (const float*)d_in[6];
    const float* g1_b1 = (const float*)d_in[7];
    const float* g1_w2 = (const float*)d_in[8];
    const float* g1_b2 = (const float*)d_in[9];
    const float* g1_w3 = (const float*)d_in[10];
    const float* g1_b3 = (const float*)d_in[11];
    const float* f1_w  = (const float*)d_in[12];
    const float* f1_b  = (const float*)d_in[13];
    const float* g2_w1 = (const float*)d_in[14];
    const float* g2_b1 = (const float*)d_in[15];
    const float* g2_w2 = (const float*)d_in[16];
    const float* g2_b2 = (const float*)d_in[17];
    const float* g2_w3 = (const float*)d_in[18];
    const float* g2_b3 = (const float*)d_in[19];
    const float* f2_w  = (const float*)d_in[20];
    const float* f2_b  = (const float*)d_in[21];
    const float* fg_w1 = (const float*)d_in[22];
    const float* fg_b1 = (const float*)d_in[23];
    const float* fg_w2 = (const float*)d_in[24];
    const float* fg_b2 = (const float*)d_in[25];
    const float* pool_coords = (const float*)d_in[26];

    char* ws = (char*)d_ws;
    unsigned short* wp = (unsigned short*)(ws + OFF_WP1);
    float* A1  = (float*)(ws + OFF_A1);
    float* B1  = (float*)(ws + OFF_B1);
    float* A2  = (float*)(ws + OFF_A2);
    float* B2P = (float*)(ws + OFF_B2P);
    float* HQ1 = (float*)(ws + OFF_HQ1);
    float* HQ2 = (float*)(ws + OFF_HQ2);
    float* S1  = (float*)(ws + OFF_S1);
    float* S2  = (float*)(ws + OFF_S2);
    float* RNO = (float*)(ws + OFF_RNO);
    float* RNOB= (float*)(ws + OFF_RNOB);
    float* Z   = (float*)(ws + OFF_Z);
    float* out = (float*)d_out;

    // 1) weight packing (bf16 WMMA-B layout)
    pack_w_kernel<<<16384, 256, 0, stream>>>(g1_w2, g1_w3, g2_w2, g2_w3, wp);

    // 2) projections
    proj_kernel<<<BATCH * NMAX, 256, 0, stream>>>(
        box_feats, box_coords, g1_w1,                         A1, NMAX, 1);     // bfc @ W1a
    proj_kernel<<<BATCH * NMAX, 256, 0, stream>>>(
        box_feats, box_coords, g1_w1 + (size_t)D_BOX * LIN,   B1, NMAX, 1);     // bfc @ W1b
    proj_kernel<<<BATCH * NMAX, 256, 0, stream>>>(
        box_feats, box_coords, g2_w1,                         A2, NMAX, 1);     // bfc @ W2a
    proj_kernel<<<BATCH * NPOOL, 256, 0, stream>>>(
        pooled, pool_coords, g2_w1 + (size_t)D_BOX * LIN,     B2P, NPOOL, 0);   // pc @ W2b

    // hq1 = q @ W1q + g1_b1 ; hq2 = q @ W2q + g2_b1
    gemv_kernel<<<BATCH * 4, 256, 0, stream>>>(
        q_feats, g1_w1 + (size_t)(2 * D_BOX) * LIN, g1_b1, HQ1, 0);
    gemv_kernel<<<BATCH * 4, 256, 0, stream>>>(
        q_feats, g2_w1 + (size_t)(2 * D_BOX) * LIN, g2_b1, HQ2, 0);

    // 3) zero accumulators (S1,S2 contiguous)
    zero_kernel<<<(2 * BATCH * LIN + 255) / 256, 256, 0, stream>>>(S1, 2 * BATCH * LIN);

    // 4) fused relational MLP (WMMA hot loop)
    const size_t shmem = (size_t)2 * PROWS * H0STR * sizeof(unsigned short)
                       + PROWS * sizeof(float);              // 132224 B
    {   // branch 1: 128x128 pairs -> 512 tiles/sample
        dim3 grid((NMAX * NMAX) / PROWS, BATCH);
        pair_kernel<<<grid, PTHREADS, shmem, stream>>>(
            A1, B1, HQ1,
            (unsigned short*)(ws + OFF_WP1), g1_b2,
            (unsigned short*)(ws + OFF_WP2), g1_b3,
            index, /*ncols_shift=*/7, /*check_j=*/1, S1);
    }
    {   // branch 2: 128x16 pairs -> 64 tiles/sample
        dim3 grid((NMAX * NPOOL) / PROWS, BATCH);
        pair_kernel<<<grid, PTHREADS, shmem, stream>>>(
            A2, B2P, HQ2,
            (unsigned short*)(ws + OFF_WP3), g2_b2,
            (unsigned short*)(ws + OFF_WP4), g2_b3,
            index, /*ncols_shift=*/4, /*check_j=*/0, S2);
    }

    // 5) heads
    gemv_kernel<<<BATCH * 4, 256, 0, stream>>>(S1, f1_w, f1_b, RNO, 1);
    gemv_kernel<<<BATCH * 4, 256, 0, stream>>>(S2, f2_w, f2_b, RNOB, 1);
    finalz_kernel<<<BATCH * 4, 256, 0, stream>>>(RNO, RNOB, fg_w1, fg_b1, Z);
    finalout_kernel<<<BATCH, 256, 0, stream>>>(Z, fg_w2, fg_b2, out);
}
